// CrossModalAttention_39195871543663
// MI455X (gfx1250) — matrix operand order repaired and checked by reading, not compiled, not run on hardware
//
#include <hip/hip_runtime.h>
#include <hip/hip_bf16.h>

typedef __bf16 bf16_t;
typedef __attribute__((ext_vector_type(16))) __bf16 v16bf;
typedef __attribute__((ext_vector_type(8)))  __bf16 v8bf;
typedef __attribute__((ext_vector_type(8)))  float  v8f;

#define B_   4
#define C_   256
#define RD_  64
#define N_   4096
#define SCALE_ 0.125f
#define EPS_   1e-5f

// workspace layout (bytes)
#define OFF_XT0   ((size_t)0)          // bf16 (B,N,C)  wli   : 8 MB
#define OFF_XT1   ((size_t)8388608)    // bf16 (B,N,C)  nbi   : 8 MB
#define OFF_Q     ((size_t)16777216)   // bf16 [2][B][N][64]   : 4 MB
#define OFF_K     ((size_t)20971520)   // bf16 [2][B][N][64]   : 4 MB
#define OFF_V     ((size_t)25165824)   // bf16 [2][B][256][N]  : 16 MB
#define OFF_STATS ((size_t)41943040)   // f32  [2][256][2]
#define OFF_WB    ((size_t)41947136)   // bf16 weights, 196608 elems

// bf16 weight sub-offsets (elements)
#define WOFF_QW0  0        // w_q_wli  (64x256)
#define WOFF_KW0  16384    // w_k_nbi  (64x256)
#define WOFF_VW0  32768    // w_v_nbi  (256x256)
#define WOFF_QW1  98304    // w_q_nbi  (64x256)
#define WOFF_KW1  114688   // w_k_wli  (64x256)
#define WOFF_VW1  131072   // w_v_wli  (256x256)
#define WTOTAL    196608

__device__ __forceinline__ bf16_t f2bf(float f) { return (bf16_t)f; }  // hw cvt
__device__ __forceinline__ unsigned short f2bf_bits(float f) {
  bf16_t h = (bf16_t)f;
  return __builtin_bit_cast(unsigned short, h);
}
__device__ __forceinline__ v8f wmma_bf16(v16bf a, v16bf b, v8f c) {
  return __builtin_amdgcn_wmma_f32_16x16x32_bf16(false, a, false, b, (short)0, c,
                                                 false, false);
}

// ---------------------------------------------------------------------------
// Kernel 0: one-shot f32 -> bf16 conversion of all six weight matrices
// ---------------------------------------------------------------------------
__global__ __launch_bounds__(256) void k_wcvt(
    const float* __restrict__ w0, const float* __restrict__ w1,
    const float* __restrict__ w2, const float* __restrict__ w3,
    const float* __restrict__ w4, const float* __restrict__ w5,
    bf16_t* __restrict__ o) {
  const int i = blockIdx.x * 256 + threadIdx.x;
  if (i >= WTOTAL) return;
  const float* src; int off;
  if      (i < WOFF_KW0) { src = w0; off = WOFF_QW0; }
  else if (i < WOFF_VW0) { src = w1; off = WOFF_KW0; }
  else if (i < WOFF_QW1) { src = w2; off = WOFF_VW0; }
  else if (i < WOFF_KW1) { src = w3; off = WOFF_QW1; }
  else if (i < WOFF_VW1) { src = w4; off = WOFF_KW1; }
  else                   { src = w5; off = WOFF_VW1; }
  o[i] = f2bf(src[i - off]);
}

// ---------------------------------------------------------------------------
// Kernel 1: f32 (B,C,N) -> bf16 (B,N,C) transpose via LDS tiles
// ---------------------------------------------------------------------------
__global__ __launch_bounds__(256) void k_pack_transpose(
    const float* __restrict__ wli, const float* __restrict__ nbi,
    bf16_t* __restrict__ xt0, bf16_t* __restrict__ xt1) {
  __shared__ float tile[32][33];
  const int md = blockIdx.z & 1;
  const int b  = blockIdx.z >> 1;
  const float* __restrict__ x = md ? nbi : wli;
  bf16_t* __restrict__ xt = md ? xt1 : xt0;
  const int n0 = blockIdx.x * 32;
  const int c0 = blockIdx.y * 32;
  const int tcol = threadIdx.x & 31;
  const int trow = threadIdx.x >> 5;
#pragma unroll
  for (int i = 0; i < 4; ++i) {
    int cl = trow + i * 8;
    tile[cl][tcol] = x[((size_t)b * C_ + (c0 + cl)) * N_ + (n0 + tcol)];
  }
  __syncthreads();
#pragma unroll
  for (int i = 0; i < 4; ++i) {
    int nl = trow + i * 8;
    xt[((size_t)b * N_ + (n0 + nl)) * C_ + (c0 + tcol)] = f2bf(tile[tcol][nl]);
  }
}

// ---------------------------------------------------------------------------
// Kernel 2: 1x1-conv projections via bf16 WMMA (bf16 weights, no inner cvt).
//   g=0 -> Q*SCALE (B,N,64), g=1 -> K (B,N,64), g=2..5 -> V (B,256,N).
// ---------------------------------------------------------------------------
__global__ __launch_bounds__(128) void k_proj(
    const bf16_t* __restrict__ xt0, const bf16_t* __restrict__ xt1,
    const bf16_t* __restrict__ Wb,
    bf16_t* __restrict__ Qws, bf16_t* __restrict__ Kws,
    bf16_t* __restrict__ Vws) {
  const int lane  = threadIdx.x & 31;
  const int wv    = threadIdx.x >> 5;
  const int khalf = lane >> 4;
  const int l16   = lane & 15;
  const int g = blockIdx.y;
  const int m = blockIdx.z & 1;
  const int b = blockIdx.z >> 1;
  const int n0 = blockIdx.x * 64 + wv * 16;

  const bf16_t* src;
  const bf16_t* W;
  int colbase = 0;
  if (m == 0) {
    if (g == 0)      { src = xt0; W = Wb + WOFF_QW0; }
    else if (g == 1) { src = xt1; W = Wb + WOFF_KW0; }
    else             { src = xt1; W = Wb + WOFF_VW0; colbase = (g - 2) * 64; }
  } else {
    if (g == 0)      { src = xt1; W = Wb + WOFF_QW1; }
    else if (g == 1) { src = xt0; W = Wb + WOFF_KW1; }
    else             { src = xt0; W = Wb + WOFF_VW1; colbase = (g - 2) * 64; }
  }

  v8f acc[4];
#pragma unroll
  for (int s = 0; s < 4; ++s)
#pragma unroll
    for (int e = 0; e < 8; ++e) acc[s][e] = 0.f;

  const bf16_t* arow = src + ((size_t)b * N_ + (n0 + l16)) * C_;

  for (int c0 = 0; c0 < C_; c0 += 32) {
    v8bf alo = *(const v8bf*)(arow + c0 + khalf * 16);
    v8bf ahi = *(const v8bf*)(arow + c0 + khalf * 16 + 8);
    v16bf A;
#pragma unroll
    for (int i = 0; i < 8; ++i) { A[i] = alo[i]; A[8 + i] = ahi[i]; }
#pragma unroll
    for (int s = 0; s < 4; ++s) {
      const int r = colbase + s * 16 + l16;
      v16bf Bv = *(const v16bf*)(W + (size_t)r * C_ + c0 + khalf * 16);
      acc[s] = wmma_bf16(A, Bv, acc[s]);
    }
  }

  if (g < 2) {
    const float sc = (g == 0) ? SCALE_ : 1.0f;  // fold 1/sqrt(rdim) into Q
    bf16_t* out = (g == 0 ? Qws : Kws) + (size_t)(m * B_ + b) * N_ * RD_;
#pragma unroll
    for (int s = 0; s < 4; ++s)
#pragma unroll
      for (int e = 0; e < 8; ++e) {
        const int nn = n0 + e + 8 * khalf;
        out[(size_t)nn * RD_ + s * 16 + l16] = f2bf(acc[s][e] * sc);
      }
  } else {
    bf16_t* out = Vws + (size_t)(m * B_ + b) * (size_t)C_ * N_;
#pragma unroll
    for (int s = 0; s < 4; ++s) {
      const int c = colbase + s * 16 + l16;
      v8bf pk;
#pragma unroll
      for (int e = 0; e < 8; ++e) pk[e] = f2bf(acc[s][e]);
      *(v8bf*)(out + (size_t)c * N_ + n0 + 8 * khalf) = pk;  // 16B store
    }
  }
}

// ---------------------------------------------------------------------------
// Kernel 3: flash attention + residual. One wave = 16 queries x 64 channels.
// Software-pipelined: V tiles (this iter) and K tiles (next iter) are issued
// before the softmax VALU block and the LDS fence, hiding their latency.
// ---------------------------------------------------------------------------
__global__ __launch_bounds__(256) void k_attn(
    const bf16_t* __restrict__ Qws, const bf16_t* __restrict__ Kws,
    const bf16_t* __restrict__ Vws,
    const float* __restrict__ wli, const float* __restrict__ nbi,
    float* __restrict__ out) {
  __shared__ alignas(32) unsigned short plds[8][512];  // 16x32 bf16 per wave
  const int lane  = threadIdx.x & 31;
  const int wslot = threadIdx.x >> 5;
  const int khalf = lane >> 4;
  const int l16   = lane & 15;

  const int wid = blockIdx.x * 8 + wslot;
  const int cg = wid & 3;           // channel group (64 ch)
  const int rb = (wid >> 2) & 255;  // 16-query row block
  const int b  = (wid >> 10) & 3;
  const int m  = wid >> 12;

  const bf16_t* Qb = Qws + (size_t)(m * B_ + b) * N_ * RD_;
  const bf16_t* Kb = Kws + (size_t)(m * B_ + b) * N_ * RD_;
  const bf16_t* Vb = Vws + (size_t)(m * B_ + b) * (size_t)C_ * N_;
  const float*  xr = (m == 0) ? wli : nbi;

  // Q A-tiles (16x32 each) for K-dim 0..31 and 32..63
  v16bf A0, A1;
  {
    const bf16_t* qrow = Qb + (size_t)(rb * 16 + l16) * RD_;
    v8bf q0 = *(const v8bf*)(qrow + khalf * 16);
    v8bf q1 = *(const v8bf*)(qrow + khalf * 16 + 8);
    v8bf q2 = *(const v8bf*)(qrow + 32 + khalf * 16);
    v8bf q3 = *(const v8bf*)(qrow + 40 + khalf * 16);
#pragma unroll
    for (int i = 0; i < 8; ++i) {
      A0[i] = q0[i]; A0[8 + i] = q1[i];
      A1[i] = q2[i]; A1[8 + i] = q3[i];
    }
  }

  v8f Oa[4];
  float mrow[8], lrow[8];
#pragma unroll
  for (int s = 0; s < 4; ++s)
#pragma unroll
    for (int e = 0; e < 8; ++e) Oa[s][e] = 0.f;
#pragma unroll
  for (int e = 0; e < 8; ++e) { mrow[e] = -1e30f; lrow[e] = 0.f; }

  unsigned short* pl = plds[wslot];

  auto loadK = [&](int p0, v16bf* Bk) {
    const bf16_t* k0 = Kb + (size_t)(p0 + l16) * RD_;
    const bf16_t* k1 = Kb + (size_t)(p0 + 16 + l16) * RD_;
    Bk[0] = *(const v16bf*)(k0 + khalf * 16);
    Bk[1] = *(const v16bf*)(k0 + 32 + khalf * 16);
    Bk[2] = *(const v16bf*)(k1 + khalf * 16);
    Bk[3] = *(const v16bf*)(k1 + 32 + khalf * 16);
  };

  v16bf Bk[4];
  loadK(0, Bk);

#pragma unroll 2
  for (int p0 = 0; p0 < N_; p0 += 32) {
    // S = (Q*scale) K^T for 16 queries x 32 keys
    v8f S0, S1;
#pragma unroll
    for (int e = 0; e < 8; ++e) { S0[e] = 0.f; S1[e] = 0.f; }
    S0 = wmma_bf16(A0, Bk[0], S0);
    S0 = wmma_bf16(A1, Bk[1], S0);
    S1 = wmma_bf16(A0, Bk[2], S1);
    S1 = wmma_bf16(A1, Bk[3], S1);

    // Early-issue: V tiles for this iteration (consumed after softmax)
    v16bf Bv[4];
#pragma unroll
    for (int s = 0; s < 4; ++s) {
      const int c = cg * 64 + s * 16 + l16;
      Bv[s] = *(const v16bf*)(Vb + (size_t)c * N_ + p0 + khalf * 16);
    }
    // Early-issue: K tiles for next iteration (wrapped, always valid)
    v16bf Bk2[4];
    loadK((p0 + 32) & (N_ - 1), Bk2);
    // L2->L0 warm-up two iterations ahead
    {
      const int pf = (p0 + 64) & (N_ - 1);
      __builtin_prefetch(Kb + (size_t)(pf + l16) * RD_, 0, 1);
      __builtin_prefetch(Vb + (size_t)(cg * 64 + l16) * N_ + pf, 0, 1);
    }

    // Online softmax: rows live across the 16-lane halves of the wave.
#pragma unroll
    for (int e = 0; e < 8; ++e) {
      const float s0 = S0[e];
      const float s1 = S1[e];
      float t = fmaxf(s0, s1);
      t = fmaxf(t, __shfl_xor(t, 1, 32));
      t = fmaxf(t, __shfl_xor(t, 2, 32));
      t = fmaxf(t, __shfl_xor(t, 4, 32));
      t = fmaxf(t, __shfl_xor(t, 8, 32));
      const float mn = fmaxf(mrow[e], t);
      const float al = __expf(mrow[e] - mn);
      mrow[e] = mn;
      const float e0 = __expf(s0 - mn);
      const float e1 = __expf(s1 - mn);
      S0[e] = e0; S1[e] = e1;
      float rs = e0 + e1;
      rs += __shfl_xor(rs, 1, 32);
      rs += __shfl_xor(rs, 2, 32);
      rs += __shfl_xor(rs, 4, 32);
      rs += __shfl_xor(rs, 8, 32);
      lrow[e] = lrow[e] * al + rs;
#pragma unroll
      for (int s = 0; s < 4; ++s) Oa[s][e] *= al;
    }

    // C-layout -> A-layout transpose of P through per-wave LDS tile
#pragma unroll
    for (int e = 0; e < 8; ++e) {
      const int row = e + 8 * khalf;
      pl[row * 32 + l16]      = f2bf_bits(S0[e]);
      pl[row * 32 + 16 + l16] = f2bf_bits(S1[e]);
    }
    asm volatile("s_wait_dscnt 0" ::: "memory");
    v16bf PA;
    {
      const unsigned short* prow = pl + l16 * 32;
      v8bf plo = *(const v8bf*)(const void*)(prow + khalf * 16);
      v8bf phi = *(const v8bf*)(const void*)(prow + khalf * 16 + 8);
#pragma unroll
      for (int i = 0; i < 8; ++i) { PA[i] = plo[i]; PA[8 + i] = phi[i]; }
    }

    // O += P V  (V tiles already in registers)
#pragma unroll
    for (int s = 0; s < 4; ++s) Oa[s] = wmma_bf16(PA, Bv[s], Oa[s]);

    // rotate pipelined K tiles
#pragma unroll
    for (int i = 0; i < 4; ++i) Bk[i] = Bk2[i];
  }

  float invl[8];
#pragma unroll
  for (int e = 0; e < 8; ++e) invl[e] = 1.0f / lrow[e];

  float* outp = out + (size_t)(m * B_ + b) * (size_t)C_ * N_;
#pragma unroll
  for (int s = 0; s < 4; ++s) {
    const int c = cg * 64 + s * 16 + l16;
    const size_t off = (size_t)(rb * 16 + 8 * khalf);
    const float* xp = xr + ((size_t)b * C_ + c) * N_ + off;
    float* op = outp + (size_t)c * N_ + off;
#pragma unroll
    for (int e = 0; e < 8; ++e) op[e] = xp[e] + Oa[s][e] * invl[e];
  }
}

// ---------------------------------------------------------------------------
// Kernel 4: per-channel BN statistics (mean, population var)
// ---------------------------------------------------------------------------
__global__ __launch_bounds__(256) void k_bnstats(const float* __restrict__ out,
                                                 float* __restrict__ stats) {
  __shared__ float sh[256], sh2[256];
  const int ch = blockIdx.x & 255;
  const int m  = blockIdx.x >> 8;
  const float* base = out + (size_t)m * B_ * C_ * N_ + (size_t)ch * N_;
  float s = 0.f, s2 = 0.f;
  for (int b = 0; b < B_; ++b) {
    const float* p = base + (size_t)b * C_ * N_;
    for (int i = threadIdx.x; i < N_; i += 256) {
      const float v = p[i];
      s += v; s2 += v * v;
    }
  }
  sh[threadIdx.x] = s; sh2[threadIdx.x] = s2;
  __syncthreads();
  for (int off = 128; off > 0; off >>= 1) {
    if ((int)threadIdx.x < off) {
      sh[threadIdx.x]  += sh[threadIdx.x + off];
      sh2[threadIdx.x] += sh2[threadIdx.x + off];
    }
    __syncthreads();
  }
  if (threadIdx.x == 0) {
    const float inv = 1.0f / (float)(B_ * N_);
    const float mean = sh[0] * inv;
    const float var  = sh2[0] * inv - mean * mean;
    stats[(m * C_ + ch) * 2 + 0] = mean;
    stats[(m * C_ + ch) * 2 + 1] = var;
  }
}

// ---------------------------------------------------------------------------
// Kernel 5: BN normalize in place
// ---------------------------------------------------------------------------
__global__ __launch_bounds__(256) void k_bnapply(
    float* __restrict__ out, const float* __restrict__ stats,
    const float* __restrict__ g0, const float* __restrict__ be0,
    const float* __restrict__ g1, const float* __restrict__ be1) {
  const size_t MS = (size_t)B_ * C_ * N_;
  const size_t i = (size_t)blockIdx.x * 256 + threadIdx.x;
  if (i >= 2 * MS) return;
  const int m = (int)(i / MS);
  const size_t r = i - (size_t)m * MS;
  const int ch = (int)((r / N_) & (C_ - 1));
  const float mean = stats[(m * C_ + ch) * 2 + 0];
  const float var  = stats[(m * C_ + ch) * 2 + 1];
  const float g  = m ? g1[ch] : g0[ch];
  const float be = m ? be1[ch] : be0[ch];
  out[i] = (out[i] - mean) * rsqrtf(var + EPS_) * g + be;
}

// ---------------------------------------------------------------------------
extern "C" void kernel_launch(void* const* d_in, const int* in_sizes, int n_in,
                              void* d_out, int out_size, void* d_ws,
                              size_t ws_size, hipStream_t stream) {
  (void)in_sizes; (void)n_in; (void)out_size; (void)ws_size;
  const float* wli    = (const float*)d_in[0];
  const float* nbi    = (const float*)d_in[1];
  const float* wq_wli = (const float*)d_in[2];
  const float* wk_nbi = (const float*)d_in[3];
  const float* wv_nbi = (const float*)d_in[4];
  const float* wq_nbi = (const float*)d_in[5];
  const float* wk_wli = (const float*)d_in[6];
  const float* wv_wli = (const float*)d_in[7];
  const float* g_wli  = (const float*)d_in[8];
  const float* b_wli  = (const float*)d_in[9];
  const float* g_nbi  = (const float*)d_in[10];
  const float* b_nbi  = (const float*)d_in[11];

  char* ws = (char*)d_ws;
  bf16_t* xt0 = (bf16_t*)(ws + OFF_XT0);
  bf16_t* xt1 = (bf16_t*)(ws + OFF_XT1);
  bf16_t* Q   = (bf16_t*)(ws + OFF_Q);
  bf16_t* K   = (bf16_t*)(ws + OFF_K);
  bf16_t* V   = (bf16_t*)(ws + OFF_V);
  float*  st  = (float*)(ws + OFF_STATS);
  bf16_t* Wb  = (bf16_t*)(ws + OFF_WB);
  float*  outp = (float*)d_out;

  k_wcvt<<<dim3((WTOTAL + 255) / 256), 256, 0, stream>>>(
      wq_wli, wk_nbi, wv_nbi, wq_nbi, wk_wli, wv_wli, Wb);
  k_pack_transpose<<<dim3(N_ / 32, C_ / 32, B_ * 2), 256, 0, stream>>>(
      wli, nbi, xt0, xt1);
  k_proj<<<dim3(N_ / 64, 6, B_ * 2), 128, 0, stream>>>(
      xt0, xt1, Wb, Q, K, V);
  k_attn<<<dim3(1024), 256, 0, stream>>>(Q, K, V, wli, nbi, outp);
  k_bnstats<<<dim3(512), 256, 0, stream>>>(outp, st);
  const size_t total = 2ull * B_ * C_ * N_;
  k_bnapply<<<dim3((unsigned)((total + 255) / 256)), 256, 0, stream>>>(
      outp, st, g_wli, b_wli, g_nbi, b_nbi);
}
